// NodeModel_35304631174017
// MI455X (gfx1250) — matrix operand order repaired and checked by reading, not compiled, not run on hardware
//
#include <hip/hip_runtime.h>

typedef __attribute__((ext_vector_type(16))) _Float16 v16h;
typedef __attribute__((ext_vector_type(8)))  float    v8f;

#define N_NODES 50000
#define N_EDGES 800000
#define IN_CH   64
#define HID     128
#define LAT     64
#define U_DIM   32
#define OUT_W   320          // 64 + 64 + 64 + 64 + 64
#define TILE_E  128

// ---- order-preserving float <-> uint encoding (for atomic max/min on u32) ----
__device__ __forceinline__ unsigned encf(float f) {
    unsigned u = __float_as_uint(f);
    return (u & 0x80000000u) ? ~u : (u | 0x80000000u);
}
__device__ __forceinline__ float decf(unsigned e) {
    return (e & 0x80000000u) ? __uint_as_float(e & 0x7FFFFFFFu) : __uint_as_float(~e);
}

// ---- WMMA fragment loaders ----
// A (16-bit, 16x32): lane<16 holds row M=lane, K = k0+0..15;
// lane>=16 holds K = k0+16..31.  Source: row-major f16 LDS tile [rows][HID].
__device__ __forceinline__ v16h load_a_frag(const _Float16* buf, int row, int k0, int hi) {
    union { v16h h; uint4 q[2]; } u;
    const _Float16* base = buf + row * HID + k0 + hi * 16;
    u.q[0] = *(const uint4*)(base);        // K chunk 0 (8 halves, 16B aligned)
    u.q[1] = *(const uint4*)(base + 8);    // K chunk +8
    return u.h;
}
// B (16-bit, 32x16): lane<16 = column n, K k0..k0+15; lane>=16 = column n, K k0+16..k0+31.
// WT is the transposed weight, row-major [N][K] f16, K == HID -> one contiguous 32B load.
__device__ __forceinline__ v16h load_b_frag(const _Float16* __restrict__ WT, int ncol, int k0, int hi) {
    return *(const v16h*)(WT + (size_t)ncol * HID + k0 + hi * 16);
}

// One 128x128 dense layer with bias+relu, in-place on the wave's own 16-row strip.
__device__ __forceinline__ void dense_relu_layer(_Float16* buf,
                                                 const _Float16* __restrict__ WT,
                                                 const float* __restrict__ bias,
                                                 int m0, int lm, int hi) {
    v16h a[4];
#pragma unroll
    for (int kk = 0; kk < 4; ++kk) a[kk] = load_a_frag(buf, m0 + lm, kk * 32, hi);

    v8f acc[8];
#pragma unroll
    for (int nt = 0; nt < 8; ++nt) {
        v8f c = {};
#pragma unroll
        for (int kk = 0; kk < 4; ++kk) {
            v16h b = load_b_frag(WT, nt * 16 + lm, kk * 32, hi);
            c = __builtin_amdgcn_wmma_f32_16x16x32_f16(false, a[kk], false, b,
                                                       (short)0, c, false, false);
        }
        acc[nt] = c;
    }
    // C/D layout: lane -> col = lm, vgpr r -> row = r + 8*hi
#pragma unroll
    for (int nt = 0; nt < 8; ++nt) {
        const int col = nt * 16 + lm;
        const float bb = bias[col];
#pragma unroll
        for (int r = 0; r < 8; ++r) {
            float v = acc[nt][r] + bb;
            v = v > 0.f ? v : 0.f;
            buf[(m0 + r + 8 * hi) * HID + col] = (_Float16)v;
        }
    }
}

// ---------------- kernel 1: weights -> transposed f16 ----------------
__global__ void prep_weights_kernel(const float* __restrict__ W1, const float* __restrict__ W2,
                                    const float* __restrict__ W3,
                                    _Float16* __restrict__ W1T, _Float16* __restrict__ W2T,
                                    _Float16* __restrict__ W3T) {
    int i = blockIdx.x * blockDim.x + threadIdx.x;
    if (i < HID * HID) {
        int o = i / HID, k = i % HID;            // W[k][o] row-major [K][N]
        W1T[i] = (_Float16)W1[k * HID + o];
        W2T[i] = (_Float16)W2[k * HID + o];
    }
    if (i < LAT * HID) {
        int o = i / HID, k = i % HID;
        W3T[i] = (_Float16)W3[k * LAT + o];
    }
}

// ---------------- kernel 2: init accumulators ----------------
__global__ void init_accum_kernel(float* __restrict__ sum, unsigned* __restrict__ mxe,
                                  unsigned* __restrict__ mne, float* __restrict__ cnt) {
    int i = blockIdx.x * blockDim.x + threadIdx.x;
    if (i < N_NODES * LAT) {
        sum[i] = 0.f;
        mxe[i] = 0u;           // encodes -NaN: below every real float
        mne[i] = 0xFFFFFFFFu;  // encodes +NaN: above every real float
    }
    if (i < N_NODES) cnt[i] = 0.f;
}

// ---------------- kernel 3: edge MLP (WMMA) + scatter ----------------
__global__ void __launch_bounds__(256)
edge_mlp_kernel(const float* __restrict__ x, const int* __restrict__ eidx,
                const _Float16* __restrict__ W1T, const float* __restrict__ b1,
                const _Float16* __restrict__ W2T, const float* __restrict__ b2,
                const _Float16* __restrict__ W3T, const float* __restrict__ b3,
                float* __restrict__ sum, unsigned* __restrict__ mxe,
                unsigned* __restrict__ mne, float* __restrict__ cnt) {
    __shared__ __align__(32) _Float16 buf[TILE_E * HID]; // 32 KB, reused across layers
    __shared__ int dcol[TILE_E];

    const int tid = threadIdx.x;
    const int e0  = blockIdx.x * TILE_E;

    // ---- gather x[row] | x[col] -> f16 tile; fold in segment count ----
    {
        const int el = tid >> 1, part = tid & 1;   // part 0: row endpoint, 1: col endpoint
        const int e  = e0 + el;
        const bool valid = (e < N_EDGES);
        int node = 0;
        if (valid) node = eidx[(size_t)part * N_EDGES + e];
        if (part) {
            dcol[el] = valid ? node : -1;
            if (valid) atomicAdd(cnt + node, 1.0f);
        }
        const float4* src = (const float4*)(x + (size_t)node * IN_CH);
        _Float16* dst = buf + el * HID + part * IN_CH;
#pragma unroll
        for (int i = 0; i < IN_CH / 4; ++i) {
            float4 f = src[i];
            union { _Float16 h[4]; uint2 u; } pk;
            pk.h[0] = (_Float16)f.x; pk.h[1] = (_Float16)f.y;
            pk.h[2] = (_Float16)f.z; pk.h[3] = (_Float16)f.w;
            *(uint2*)(dst + 4 * i) = pk.u;
        }
    }
    __syncthreads();

    const int wave = tid >> 5, lane = tid & 31;
    const int hi = lane >> 4, lm = lane & 15;
    const int m0 = wave * 16;                  // this wave's 16-edge strip

    dense_relu_layer(buf, W1T, b1, m0, lm, hi);
    __syncthreads();
    dense_relu_layer(buf, W2T, b2, m0, lm, hi);
    __syncthreads();

    // ---- layer 3: 128x128 @ 128x64 -> registers -> atomic scatter ----
    {
        v16h a[4];
#pragma unroll
        for (int kk = 0; kk < 4; ++kk) a[kk] = load_a_frag(buf, m0 + lm, kk * 32, hi);
#pragma unroll
        for (int nt = 0; nt < 4; ++nt) {
            v8f c = {};
#pragma unroll
            for (int kk = 0; kk < 4; ++kk) {
                v16h b = load_b_frag(W3T, nt * 16 + lm, kk * 32, hi);
                c = __builtin_amdgcn_wmma_f32_16x16x32_f16(false, a[kk], false, b,
                                                           (short)0, c, false, false);
            }
            const int ch = nt * 16 + lm;
            const float bb = b3[ch];
#pragma unroll
            for (int r = 0; r < 8; ++r) {
                const int node = dcol[m0 + r + 8 * hi];
                if (node >= 0) {
                    const float v = c[r] + bb;
                    const size_t o = (size_t)node * LAT + ch;
                    atomicAdd(sum + o, v);
                    atomicMax(mxe + o, encf(v));
                    atomicMin(mne + o, encf(v));
                }
            }
        }
    }
}

// ---------------- kernel 4: finalize / concat ----------------
__global__ void finalize_kernel(const float* __restrict__ x, const float* __restrict__ u,
                                const int* __restrict__ batch,
                                const float* __restrict__ sum, const unsigned* __restrict__ mxe,
                                const unsigned* __restrict__ mne, const float* __restrict__ cnt,
                                float* __restrict__ out) {
    long long i = (long long)blockIdx.x * blockDim.x + threadIdx.x;
    if (i >= (long long)N_NODES * OUT_W) return;
    const int n = (int)(i / OUT_W);
    const int c = (int)(i % OUT_W);
    float v;
    if (c < 64) {
        v = x[(size_t)n * IN_CH + c];
    } else if (c < 128) {
        v = sum[(size_t)n * LAT + (c - 64)] / fmaxf(cnt[n], 1.0f);
    } else if (c < 192) {
        v = (cnt[n] > 0.f) ? decf(mxe[(size_t)n * LAT + (c - 128)]) : 0.f;
    } else if (c < 256) {
        v = (cnt[n] > 0.f) ? decf(mne[(size_t)n * LAT + (c - 192)]) : 0.f;
    } else {
        v = u[(size_t)batch[n] * U_DIM + (c - 256)];
    }
    out[i] = v;
}

extern "C" void kernel_launch(void* const* d_in, const int* in_sizes, int n_in,
                              void* d_out, int out_size, void* d_ws, size_t ws_size,
                              hipStream_t stream) {
    const float* x     = (const float*)d_in[0];
    const int*   eidx  = (const int*)d_in[1];
    // d_in[2] = edge_attr (unused by the reference NodeModel)
    const float* u     = (const float*)d_in[3];
    const int*   batch = (const int*)d_in[4];
    const float* W1 = (const float*)d_in[5];
    const float* b1 = (const float*)d_in[6];
    const float* W2 = (const float*)d_in[7];
    const float* b2 = (const float*)d_in[8];
    const float* W3 = (const float*)d_in[9];
    const float* b3 = (const float*)d_in[10];
    float* out = (float*)d_out;

    // workspace layout (all 256B-aligned slabs)
    char* p = (char*)d_ws;
    _Float16* W1T = (_Float16*)p; p += (size_t)HID * HID * 2;
    _Float16* W2T = (_Float16*)p; p += (size_t)HID * HID * 2;
    _Float16* W3T = (_Float16*)p; p += (size_t)LAT * HID * 2;
    float*    sum = (float*)p;    p += (size_t)N_NODES * LAT * 4;
    unsigned* mxe = (unsigned*)p; p += (size_t)N_NODES * LAT * 4;
    unsigned* mne = (unsigned*)p; p += (size_t)N_NODES * LAT * 4;
    float*    cnt = (float*)p;    p += (size_t)N_NODES * 4;

    prep_weights_kernel<<<(HID * HID + 255) / 256, 256, 0, stream>>>(W1, W2, W3, W1T, W2T, W3T);
    init_accum_kernel<<<(N_NODES * LAT + 255) / 256, 256, 0, stream>>>(sum, mxe, mne, cnt);
    edge_mlp_kernel<<<(N_EDGES + TILE_E - 1) / TILE_E, 256, 0, stream>>>(
        x, eidx, W1T, b1, W2T, b2, W3T, b3, sum, mxe, mne, cnt);
    finalize_kernel<<<(int)(((long long)N_NODES * OUT_W + 255) / 256), 256, 0, stream>>>(
        x, u, batch, sum, mxe, mne, cnt, out);
}
